// AQLMQuantizedLoRA_5781025980674
// MI455X (gfx1250) — matrix operand order repaired and checked by reading, not compiled, not run on hardware
//
#include <hip/hip_runtime.h>
#include <hip/hip_bf16.h>

// ---------------------------------------------------------------------------
// AQLM quantized linear + folded LoRA for MI455X (gfx1250, wave32, WMMA).
//   out[m, o] = sum_k xbf16[m,k] * Weff_bf16[o,k] + bias[o]
//   Weff = dequant(codes, codebooks, scales) + (alpha/r) * lora_B @ lora_A
// GEMM uses v_wmma_f32_16x16x32_bf16 with GLOBAL_LOAD_ASYNC_TO_LDS_B128
// staging (ASYNCcnt) — no VGPR round-trip for tile data.
// ---------------------------------------------------------------------------

typedef __attribute__((ext_vector_type(16))) __bf16 v16bf;
typedef __attribute__((ext_vector_type(8)))  __bf16 v8bf;
typedef __attribute__((ext_vector_type(8)))  float  v8f;

#define M_TOT   8192      // 4 * 2048 tokens
#define K_TOT   4096      // IN_FEATURES
#define N_TOT   4096      // OUT_FEATURES
#define NGROUPS 512       // IN_FEATURES / IN_GROUP
#define LORA_SCALE 2.0f   // alpha/r = 32/16

// ---------------------------------------------------------------------------
// gfx1250 async copy: global -> LDS, 16 bytes per lane, tracked by ASYNCcnt.
// VDST field carries the LDS byte address; per the CDNA5 aperture rules the
// LDS aperture is encoded in addr[63:32], so addr[31:0] of a generic pointer
// to __shared__ data is exactly the LDS byte offset.
// ---------------------------------------------------------------------------
__device__ __forceinline__ void async_copy_b128(__bf16* lds_dst, const __bf16* gsrc) {
    unsigned loff = (unsigned)(unsigned long long)lds_dst;
    unsigned long long gaddr = (unsigned long long)gsrc;
    asm volatile("global_load_async_to_lds_b128 %0, %1, off"
                 :: "v"(loff), "v"(gaddr)
                 : "memory");
}

__device__ __forceinline__ void wait_async_done() {
#if __has_builtin(__builtin_amdgcn_s_wait_asynccnt)
    __builtin_amdgcn_s_wait_asynccnt(0);
#else
    asm volatile("s_wait_asynccnt 0x0" ::: "memory");
#endif
}

// ---------------------------------------------------------------------------
// Pass 1: x fp32 -> bf16 (8 elements per thread, exact grid coverage)
// ---------------------------------------------------------------------------
__global__ __launch_bounds__(256) void convert_x_kernel(const float* __restrict__ x,
                                                        unsigned short* __restrict__ xb_raw) {
    __bf16* xb = reinterpret_cast<__bf16*>(xb_raw);
    size_t i = ((size_t)blockIdx.x * 256 + threadIdx.x) * 8;
    float4 f0 = *(const float4*)(x + i);
    float4 f1 = *(const float4*)(x + i + 4);
    v8bf v;
    v[0] = (__bf16)f0.x; v[1] = (__bf16)f0.y; v[2] = (__bf16)f0.z; v[3] = (__bf16)f0.w;
    v[4] = (__bf16)f1.x; v[5] = (__bf16)f1.y; v[6] = (__bf16)f1.z; v[7] = (__bf16)f1.w;
    *(v8bf*)(xb + i) = v;
}

// ---------------------------------------------------------------------------
// Pass 2: dequantize AQLM codes + fold rank-16 LoRA into Weff (bf16).
// One thread per (o, group): 8 contiguous K elements.
// ---------------------------------------------------------------------------
__global__ __launch_bounds__(256) void dequant_fold_kernel(const int*   __restrict__ codes,
                                                           const float* __restrict__ codebooks,
                                                           const float* __restrict__ scales,
                                                           const float* __restrict__ lora_A,
                                                           const float* __restrict__ lora_B,
                                                           unsigned short* __restrict__ wq_raw) {
    __bf16* wq = reinterpret_cast<__bf16*>(wq_raw);
    int gid = blockIdx.x * 256 + threadIdx.x;      // 0 .. N_TOT*NGROUPS-1
    int o = gid >> 9;                              // / NGROUPS
    int g = gid & (NGROUPS - 1);

    int2 cp = ((const int2*)codes)[(size_t)o * NGROUPS + g];   // two codebook indices
    const float4* q0 = (const float4*)(codebooks + (size_t)cp.x * 8);           // codebook 0
    const float4* q1 = (const float4*)(codebooks + 2048 + (size_t)cp.y * 8);    // codebook 1
    float4 c00 = q0[0], c01 = q0[1];
    float4 c10 = q1[0], c11 = q1[1];
    float s = scales[o];

    // rank-16 LoRA contribution for these 8 K positions
    float l[8] = {0.f, 0.f, 0.f, 0.f, 0.f, 0.f, 0.f, 0.f};
    const float* Brow = lora_B + (size_t)o * 16;
    #pragma unroll
    for (int r = 0; r < 16; ++r) {
        float br = Brow[r];
        const float4* ar = (const float4*)(lora_A + (size_t)r * K_TOT + g * 8);
        float4 a0 = ar[0], a1 = ar[1];
        l[0] += br * a0.x; l[1] += br * a0.y; l[2] += br * a0.z; l[3] += br * a0.w;
        l[4] += br * a1.x; l[5] += br * a1.y; l[6] += br * a1.z; l[7] += br * a1.w;
    }

    float w0 = (c00.x + c10.x) * s + LORA_SCALE * l[0];
    float w1 = (c00.y + c10.y) * s + LORA_SCALE * l[1];
    float w2 = (c00.z + c10.z) * s + LORA_SCALE * l[2];
    float w3 = (c00.w + c10.w) * s + LORA_SCALE * l[3];
    float w4 = (c01.x + c11.x) * s + LORA_SCALE * l[4];
    float w5 = (c01.y + c11.y) * s + LORA_SCALE * l[5];
    float w6 = (c01.z + c11.z) * s + LORA_SCALE * l[6];
    float w7 = (c01.w + c11.w) * s + LORA_SCALE * l[7];

    v8bf v;
    v[0] = (__bf16)w0; v[1] = (__bf16)w1; v[2] = (__bf16)w2; v[3] = (__bf16)w3;
    v[4] = (__bf16)w4; v[5] = (__bf16)w5; v[6] = (__bf16)w6; v[7] = (__bf16)w7;
    *(v8bf*)(wq + (size_t)o * K_TOT + g * 8) = v;
}

// ---------------------------------------------------------------------------
// Pass 3: bf16 WMMA GEMM + bias epilogue.
// Block = 256 threads (8 wave32s), computes a 256(M) x 128(N) tile.
// Waves arranged 4(M) x 2(N); each wave -> 64x64 = 4x4 WMMA accumulators
// (16 v_wmma_f32_16x16x32_bf16 per K-step of 32).
// Tile staging via GLOBAL_LOAD_ASYNC_TO_LDS_B128 (ASYNCcnt), 24 KB LDS.
// ---------------------------------------------------------------------------
__global__ __launch_bounds__(256) void gemm_bias_kernel(const unsigned short* __restrict__ xb_raw,
                                                        const unsigned short* __restrict__ wq_raw,
                                                        const float* __restrict__ bias,
                                                        float* __restrict__ out) {
    const __bf16* X = reinterpret_cast<const __bf16*>(xb_raw);   // [M_TOT][K_TOT]
    const __bf16* W = reinterpret_cast<const __bf16*>(wq_raw);   // [N_TOT][K_TOT]

    __shared__ __bf16 As[256 * 32];   // [m][k] row-major, 16 KB
    __shared__ __bf16 Bs[128 * 32];   // [n][k] row-major,  8 KB

    const int tid  = threadIdx.x;
    const int lane = tid & 31;
    const int wave = tid >> 5;
    const int wm   = wave >> 1;          // 0..3  -> M offset 64*wm
    const int wn   = wave & 1;           // 0..1  -> N offset 64*wn
    const int mb   = blockIdx.y * 256;
    const int nb   = blockIdx.x * 128;

    const int half   = lane >> 4;        // 0: lanes 0-15, 1: lanes 16-31
    const int l15    = lane & 15;
    const int a_koff = half * 16;        // A frag: K 0..15 vs 16..31 (contiguous)
    const int b_koff = half * 16;        // B frag: K 0..15 vs 16..31 (contiguous)

    v8f acc[4][4];
    #pragma unroll
    for (int i = 0; i < 4; ++i)
        #pragma unroll
        for (int j = 0; j < 4; ++j)
            acc[i][j] = (v8f){0.f, 0.f, 0.f, 0.f, 0.f, 0.f, 0.f, 0.f};

    for (int k0 = 0; k0 < K_TOT; k0 += 32) {
        __syncthreads();   // previous tile fully consumed by all waves

        // A tile: 256 rows x 32 k = 1024 x 16B chunks, 4 per thread
        #pragma unroll
        for (int j = 0; j < 4; ++j) {
            int i   = tid + j * 256;
            int row = i >> 2;             // 0..255
            int kc  = (i & 3) << 3;       // 0,8,16,24
            const __bf16* src = &X[(size_t)(mb + row) * K_TOT + k0 + kc];
            async_copy_b128(&As[row * 32 + kc], src);
            __builtin_prefetch(src + 32, 0, 1);   // next K tile into L2
        }
        // B tile: 128 rows x 32 k = 512 x 16B chunks, 2 per thread
        #pragma unroll
        for (int j = 0; j < 2; ++j) {
            int i   = tid + j * 256;
            int row = i >> 2;             // 0..127
            int kc  = (i & 3) << 3;
            const __bf16* src = &W[(size_t)(nb + row) * K_TOT + k0 + kc];
            async_copy_b128(&Bs[row * 32 + kc], src);
            __builtin_prefetch(src + 32, 0, 1);
        }
        wait_async_done();   // this wave's async transfers landed in LDS
        __syncthreads();     // everyone's transfers visible

        // A fragments: lane<16 -> row M=l15, elems = K {koff..+15} (contiguous)
        v16bf a[4];
        #pragma unroll
        for (int sa = 0; sa < 4; ++sa) {
            int row = wm * 64 + sa * 16 + l15;
            v8bf lo = *(v8bf*)&As[row * 32 + a_koff];
            v8bf hi = *(v8bf*)&As[row * 32 + a_koff + 8];
            a[sa] = __builtin_shufflevector(lo, hi, 0, 1, 2, 3, 4, 5, 6, 7,
                                            8, 9, 10, 11, 12, 13, 14, 15);
        }
        // B fragments: lane<16 -> col N=l15, K 0..15; lane>=16 -> K 16..31
        v16bf b[4];
        #pragma unroll
        for (int sn = 0; sn < 4; ++sn) {
            int col = wn * 64 + sn * 16 + l15;
            b[sn] = *(v16bf*)&Bs[col * 32 + b_koff];
        }

        #pragma unroll
        for (int sa = 0; sa < 4; ++sa)
            #pragma unroll
            for (int sn = 0; sn < 4; ++sn)
                acc[sa][sn] = __builtin_amdgcn_wmma_f32_16x16x32_bf16(
                    /*neg_a=*/false, a[sa], /*neg_b=*/false, b[sn],
                    /*c_mod=*/(short)0, acc[sa][sn],
                    /*reuse_a=*/false, /*reuse_b=*/false);
    }

    // Epilogue. C/D layout: lanes 0-15: N=lane, VGPR r -> M=r; lanes 16-31: M=8+r.
    #pragma unroll
    for (int sn = 0; sn < 4; ++sn) {
        int n = nb + wn * 64 + sn * 16 + l15;
        float bn = bias[n];
        #pragma unroll
        for (int sa = 0; sa < 4; ++sa) {
            int mbase = mb + wm * 64 + sa * 16 + half * 8;
            #pragma unroll
            for (int r = 0; r < 8; ++r)
                out[(size_t)(mbase + r) * N_TOT + n] = acc[sa][sn][r] + bn;
        }
    }
}

// ---------------------------------------------------------------------------
// Host launcher
// ---------------------------------------------------------------------------
extern "C" void kernel_launch(void* const* d_in, const int* in_sizes, int n_in,
                              void* d_out, int out_size, void* d_ws, size_t ws_size,
                              hipStream_t stream) {
    (void)in_sizes; (void)n_in; (void)out_size; (void)ws_size;

    const float* x         = (const float*)d_in[0];   // [4,2048,4096] fp32
    const int*   codes     = (const int*)  d_in[1];   // [4096,512,2]
    const float* codebooks = (const float*)d_in[2];   // [2,256,1,8]
    const float* scales    = (const float*)d_in[3];   // [4096]
    const float* bias      = (const float*)d_in[4];   // [4096]
    const float* lora_A    = (const float*)d_in[5];   // [16,4096]
    const float* lora_B    = (const float*)d_in[6];   // [4096,16]
    float*       out       = (float*)d_out;           // [4,2048,4096] fp32

    // Workspace: x_bf16 (64 MB) then Weff_bf16 (32 MB) -- both L2-resident (192 MB)
    unsigned short* xb = (unsigned short*)d_ws;
    unsigned short* wq = xb + (size_t)M_TOT * K_TOT;

    // Pass 1: x -> bf16.  33,554,432 elems / 8 per thread / 256 per block.
    convert_x_kernel<<<(M_TOT * (size_t)K_TOT) / (8 * 256), 256, 0, stream>>>(x, xb);

    // Pass 2: dequant + LoRA fold.  4096*512 threads.
    dequant_fold_kernel<<<(N_TOT * NGROUPS) / 256, 256, 0, stream>>>(
        codes, codebooks, scales, lora_A, lora_B, wq);

    // Pass 3: WMMA GEMM + bias.  Grid: N/128 x M/256.
    gemm_bias_kernel<<<dim3(N_TOT / 128, M_TOT / 256), 256, 0, stream>>>(xb, wq, bias, out);
}
